// HeatTransferNetwork_28767690948634
// MI455X (gfx1250) — matrix operand order repaired and checked by reading, not compiled, not run on hardware
//
#include <hip/hip_runtime.h>
#include <hip/hip_bf16.h>
#include <stdint.h>

#define N_NODES 50000
#define N_EDGES 800000
#define HID 64
#define KC 8
#define EC_TILES (N_EDGES / 16)       // 50000 (divides exactly)
#define MK_TILES (N_EDGES / 16 + 9)   // bucket padding adds at most 9 tiles
#define PERM_CAP (MK_TILES * 16)

typedef __attribute__((ext_vector_type(16))) _Float16 v16h;
typedef __attribute__((ext_vector_type(8)))  _Float16 v8h;
typedef __attribute__((ext_vector_type(8)))  float    v8f;

// ---------------- WMMA helpers (CDNA5 16x16x32 f16 -> f32) ----------------

static __device__ __forceinline__ v8f wmma_f16(v16h a, v16h b, v8f c) {
  // (neg_a, A, neg_b, B, c_mod, C, reuse_a, reuse_b)
  return __builtin_amdgcn_wmma_f32_16x16x32_f16(false, a, false, b, (short)0, c,
                                                false, false);
}

// A (16xK, f16, row-major in LDS, row stride = `stride` elems) fragment at k0.
// Layout (matches B): lanes 0-15: M=lane, K = k0..k0+15;
//             lanes 16-31: M=lane-16, K = k0+16..k0+31.
static __device__ __forceinline__ v16h frag_a_lds(const _Float16* base, int k0,
                                                  int stride) {
  int lane = threadIdx.x & 31;
  int m = lane & 15;
  int hi = lane >> 4;
  const _Float16* row = base + m * stride + k0 + hi * 16;
  return *(const v16h*)(row);
}

// B (KxN) fragment from a row-major [out][in] f16 weight matrix (== B col-major).
// Layout (matches A): lanes 0-15: N=lane, K=k0..k0+15; lanes 16-31: N=lane-16, K=k0+16..k0+31.
static __device__ __forceinline__ v16h frag_b_rm(const _Float16* w, int n0,
                                                 int k0, int kdim) {
  int lane = threadIdx.x & 31;
  int n = n0 + (lane & 15);
  int hi = lane >> 4;
  return *(const v16h*)(w + n * kdim + k0 + hi * 16);
}

static __device__ __forceinline__ void atomicMaxF(float* a, float v) {
  if (v >= 0.f) atomicMax((int*)a, __float_as_int(v));
  else          atomicMin((unsigned int*)a, __float_as_uint(v));
}

// ---------------- utility kernels ----------------

__global__ void fill_u32_kernel(uint32_t* p, uint32_t v, int n) {
  int i = blockIdx.x * 256 + threadIdx.x;
  if (i < n) p[i] = v;
}

__global__ void f32_to_f16_kernel(const float* s, _Float16* d, int n) {
  int i = blockIdx.x * 256 + threadIdx.x;
  if (i < n) d[i] = (_Float16)s[i];
}

// xs = relu([x, pos] @ lin_sim_w^T + b), thread per (node, j)
__global__ void xs_kernel(const float* __restrict__ x, const float* __restrict__ pos,
                          const float* __restrict__ w, const float* __restrict__ b,
                          float* __restrict__ xs) {
  int i = blockIdx.x * 256 + threadIdx.x;
  if (i >= N_NODES * HID) return;
  int node = i >> 6, j = i & 63;
  const float* wr = w + j * 5;
  float v = b[j] + x[node * 3] * wr[0] + x[node * 3 + 1] * wr[1] +
            x[node * 3 + 2] * wr[2] + pos[node * 2] * wr[3] +
            pos[node * 2 + 1] * wr[4];
  xs[i] = v > 0.f ? v : 0.f;
}

// ---------------- EdgeConv: m = relu([xi, xj-xi]@w1^T+b1)@w2^T+b2; max at dst ----

__global__ void __launch_bounds__(32)
edgeconv_kernel(const float* __restrict__ xs, const int* __restrict__ src,
                const int* __restrict__ dst, const _Float16* __restrict__ w1h,
                const float* __restrict__ b1, const _Float16* __restrict__ w2h,
                const float* __restrict__ b2, float* __restrict__ agg) {
  __shared__ __align__(32) _Float16 lA[16 * 128];
  __shared__ __align__(32) _Float16 lH[16 * 64];
  __shared__ int ldst[16];
  int t = blockIdx.x;
  int lane = threadIdx.x;
  int m = lane >> 1, half = lane & 1;
  int e = t * 16 + m;
  int d = dst[e], s = src[e];
  if (half == 0) ldst[m] = d;
  const float* pd = xs + d * 64;
  const float* ps = xs + s * 64;
  _Float16* row = lA + m * 128 + half * 64;
  if (half == 0) {
#pragma unroll 8
    for (int c = 0; c < 64; c++) row[c] = (_Float16)pd[c];
  } else {
#pragma unroll 8
    for (int c = 0; c < 64; c++) row[c] = (_Float16)(ps[c] - pd[c]);
  }
  __syncthreads();
  int hi = lane >> 4, nlo = lane & 15;
  // GEMM1 (16x128)@(128x64) + bias + relu -> lH (f16)
#pragma unroll
  for (int nt = 0; nt < 4; nt++) {
    v8f acc = {};
#pragma unroll
    for (int kt = 0; kt < 4; kt++) {
      v16h a = frag_a_lds(lA, kt * 32, 128);
      v16h bf = frag_b_rm(w1h, nt * 16, kt * 32, 128);
      acc = wmma_f16(a, bf, acc);
    }
    float bias = b1[nt * 16 + nlo];
#pragma unroll
    for (int r = 0; r < 8; r++) {
      float v = acc[r] + bias;
      v = v > 0.f ? v : 0.f;
      lH[(r + 8 * hi) * 64 + nt * 16 + nlo] = (_Float16)v;
    }
  }
  __syncthreads();
  // GEMM2 (16x64)@(64x64) + bias -> atomic max into agg[dst]
#pragma unroll
  for (int nt = 0; nt < 4; nt++) {
    v8f acc = {};
#pragma unroll
    for (int kt = 0; kt < 2; kt++) {
      v16h a = frag_a_lds(lH, kt * 32, 64);
      v16h bf = frag_b_rm(w2h, nt * 16, kt * 32, 64);
      acc = wmma_f16(a, bf, acc);
    }
    float bias = b2[nt * 16 + nlo];
#pragma unroll
    for (int r = 0; r < 8; r++) {
      float v = acc[r] + bias;
      atomicMaxF(agg + (long)ldst[r + 8 * hi] * 64 + nt * 16 + nlo, v);
    }
  }
}

// xs2 = relu(agg with -inf -> 0) == max(agg, 0), in place
__global__ void fix_agg_kernel(float* agg, int n) {
  int i = blockIdx.x * 256 + threadIdx.x;
  if (i < n) agg[i] = fmaxf(agg[i], 0.f);
}

// ---------------- k-means ----------------

__global__ void init_cent_kernel(const float* __restrict__ xs2, float* cent) {
  int i = threadIdx.x;  // 512
  cent[i] = xs2[(size_t)(i >> 6) * (N_NODES / KC) * 64 + (i & 63)];
}

__global__ void km_assign_kernel(const float* __restrict__ xs2,
                                 const float* __restrict__ cent, int* assign,
                                 float* sums, float* counts) {
  __shared__ float sc[KC * 64];
  __shared__ float ps[KC * 64];
  __shared__ float pc[KC];
  int tid = threadIdx.x;
  for (int i = tid; i < KC * 64; i += 256) { sc[i] = cent[i]; ps[i] = 0.f; }
  if (tid < KC) pc[tid] = 0.f;
  __syncthreads();
  int node = blockIdx.x * 256 + tid;
  if (node < N_NODES) {
    const float* xr = xs2 + (size_t)node * 64;
    float dist[KC];
#pragma unroll
    for (int k = 0; k < KC; k++) dist[k] = 0.f;
#pragma unroll 8
    for (int d2 = 0; d2 < 64; d2++) {
      float xv = xr[d2];
#pragma unroll
      for (int k = 0; k < KC; k++) {
        float df = xv - sc[k * 64 + d2];
        dist[k] += df * df;
      }
    }
    int best = 0;
    float bd = dist[0];
#pragma unroll
    for (int k = 1; k < KC; k++)
      if (dist[k] < bd) { bd = dist[k]; best = k; }
    assign[node] = best;
#pragma unroll 8
    for (int d2 = 0; d2 < 64; d2++) atomicAdd(&ps[best * 64 + d2], xr[d2]);
    atomicAdd(&pc[best], 1.f);
  }
  __syncthreads();
  for (int i = tid; i < KC * 64; i += 256) atomicAdd(&sums[i], ps[i]);
  if (tid < KC) atomicAdd(&counts[tid], pc[tid]);
}

__global__ void km_update_kernel(const float* sums, const float* counts,
                                 float* cent) {
  int i = threadIdx.x;  // 512
  cent[i] = sums[i] / fmaxf(counts[i >> 6], 1.f);
}

// ---------------- edge prep: deg, ke, bucket counts, sort, tile k ------------

__global__ void edge_prep_kernel(const int* __restrict__ src,
                                 const int* __restrict__ dst,
                                 const int* __restrict__ assign, int* deg,
                                 int* ke, int* cnt9) {
  int e = blockIdx.x * 256 + threadIdx.x;
  if (e >= N_EDGES) return;
  int s = src[e], d = dst[e];
  atomicAdd(&deg[s], 1);
  int cs = assign[s];
  int k = (cs == assign[d]) ? cs : -1;
  ke[e] = k;
  atomicAdd(&cnt9[k < 0 ? 8 : k], 1);
}

__global__ void aoff_kernel(const int* cnt9, int* aoff) {
  if (threadIdx.x == 0 && blockIdx.x == 0) {
    int a = 0;
    for (int b = 0; b < 9; b++) {
      aoff[b] = a;
      a += (cnt9[b] + 15) & ~15;  // pad each bucket to tile multiple
    }
    aoff[9] = a;
  }
}

__global__ void scatter_kernel(const int* __restrict__ ke,
                               const int* __restrict__ aoff, int* cur9,
                               int* perm) {
  int e = blockIdx.x * 256 + threadIdx.x;
  if (e >= N_EDGES) return;
  int k = ke[e];
  int b = k < 0 ? 8 : k;
  int pos = aoff[b] + atomicAdd(&cur9[b], 1);
  perm[pos] = e;
}

__global__ void tilek_kernel(const int* __restrict__ aoff, int* tile_k) {
  int t = blockIdx.x * 256 + threadIdx.x;
  if (t >= MK_TILES) return;
  int pos = t * 16;
  int k = -1;
  if (pos < aoff[9]) {
    for (int b = 0; b < 9; b++)
      if (pos >= aoff[b] && pos < aoff[b + 1]) { k = (b == 8) ? -1 : b; break; }
  }
  tile_k[t] = k;
}

// ea0 = (pos[dst]-pos[src]) @ lin_edge_w^T + b, stored f16
__global__ void ea0_kernel(const float* __restrict__ pos,
                           const int* __restrict__ src,
                           const int* __restrict__ dst,
                           const float* __restrict__ w,
                           const float* __restrict__ b, _Float16* ea) {
  int i = blockIdx.x * 256 + threadIdx.x;
  if (i >= N_EDGES * 64) return;
  int e = i >> 6, j = i & 63;
  int s = src[e], d = dst[e];
  float dx = pos[d * 2] - pos[s * 2];
  float dy = pos[d * 2 + 1] - pos[s * 2 + 1];
  ea[i] = (_Float16)(dx * w[j * 2] + dy * w[j * 2 + 1] + b[j]);
}

// ---------------- mk_conv (HID->HID): ea_next = (sum_p f_p(ea@Wp^T) @ a[k,p]^T)/deg

__global__ void __launch_bounds__(32)
mkconv_kernel(const _Float16* __restrict__ eaPrev, _Float16* __restrict__ eaNext,
              const _Float16* __restrict__ Wh,   // 4*64*64 [p][o][c]
              const _Float16* __restrict__ Ah,   // 8*4*64*64 [k][p][q][o]
              const int* __restrict__ perm, const int* __restrict__ tile_k,
              const int* __restrict__ src, const int* __restrict__ deg) {
  __shared__ __align__(32) _Float16 lA[16 * 64];      // input tile; reused for output
  __shared__ __align__(32) _Float16 lP[4 * 16 * 64];  // power matrices
  __shared__ float sdeg[16];
  int t = blockIdx.x;
  int lane = threadIdx.x;
  int k = tile_k[t];
  int m = lane >> 1, half = lane & 1;
  int e = perm[t * 16 + m];
  if (half == 0) sdeg[m] = (e >= 0) ? (1.f / (float)deg[src[e]]) : 0.f;
  {
    _Float16* row = lA + m * 64 + half * 32;
    if (e >= 0) {
      const _Float16* g = eaPrev + (long)e * 64 + half * 32;
      *(v8h*)(row) = *(const v8h*)(g);
      *(v8h*)(row + 8) = *(const v8h*)(g + 8);
      *(v8h*)(row + 16) = *(const v8h*)(g + 16);
      *(v8h*)(row + 24) = *(const v8h*)(g + 24);
    } else {
      v8h z = {};
      *(v8h*)(row) = z; *(v8h*)(row + 8) = z;
      *(v8h*)(row + 16) = z; *(v8h*)(row + 24) = z;
    }
  }
  __syncthreads();
  if (k < 0) {  // uniform per tile: mixed-cluster / dummy edges -> zeros
    if (e >= 0) {
      _Float16* g = eaNext + (long)e * 64 + half * 32;
      v8h z = {};
      *(v8h*)(g) = z; *(v8h*)(g + 8) = z;
      *(v8h*)(g + 16) = z; *(v8h*)(g + 24) = z;
    }
    return;
  }
  int hi = lane >> 4, nlo = lane & 15;
  // GEMM1: h_p = A @ Wp^T, apply p0=h0, p_i=leaky(h_i)^i, store f16 tiles
#pragma unroll
  for (int p = 0; p < 4; p++) {
    const _Float16* wp = Wh + p * 64 * 64;
#pragma unroll
    for (int nt = 0; nt < 4; nt++) {
      v8f acc = {};
#pragma unroll
      for (int kt = 0; kt < 2; kt++) {
        v16h a = frag_a_lds(lA, kt * 32, 64);
        v16h bf = frag_b_rm(wp, nt * 16, kt * 32, 64);
        acc = wmma_f16(a, bf, acc);
      }
#pragma unroll
      for (int r = 0; r < 8; r++) {
        float v = acc[r];
        if (p > 0) {
          float lt = v >= 0.f ? v : 0.1f * v;
          v = lt;
          if (p >= 2) v *= lt;
          if (p == 3) v *= lt;
        }
        lP[p * 1024 + (r + 8 * hi) * 64 + nt * 16 + nlo] = (_Float16)v;
      }
    }
  }
  __syncthreads();
  // GEMM2: out = sum_p pmat_p @ alpha[k,p]^T, scale by 1/deg[src], store f16
  const _Float16* ak = Ah + (long)k * 4 * 64 * 64;
#pragma unroll
  for (int nt = 0; nt < 4; nt++) {
    v8f acc = {};
#pragma unroll
    for (int p = 0; p < 4; p++) {
      const _Float16* ap = ak + p * 64 * 64;
#pragma unroll
      for (int kt = 0; kt < 2; kt++) {
        v16h a = frag_a_lds(lP + p * 1024, kt * 32, 64);
        v16h bf = frag_b_rm(ap, nt * 16, kt * 32, 64);
        acc = wmma_f16(a, bf, acc);
      }
    }
#pragma unroll
    for (int r = 0; r < 8; r++) {
      int rr = r + 8 * hi;
      lA[rr * 64 + nt * 16 + nlo] = (_Float16)(acc[r] * sdeg[rr]);
    }
  }
  __syncthreads();
  if (e >= 0) {
    const _Float16* row = lA + m * 64 + half * 32;
    _Float16* g = eaNext + (long)e * 64 + half * 32;
    *(v8h*)(g) = *(const v8h*)(row);
    *(v8h*)(g + 8) = *(const v8h*)(row + 8);
    *(v8h*)(g + 16) = *(const v8h*)(row + 16);
    *(v8h*)(g + 24) = *(const v8h*)(row + 24);
  }
}

// ---------------- conv5 (HID->1) + segment_sum at dst = final output ---------

__global__ void conv5_kernel(const _Float16* __restrict__ ea,
                             const _Float16* __restrict__ W5h,  // [p][c] 4x64
                             const float* __restrict__ alpha5,  // [k][p] 8x4
                             const int* __restrict__ src,
                             const int* __restrict__ dst,
                             const int* __restrict__ ke,
                             const int* __restrict__ deg, float* out) {
  int e = blockIdx.x * 256 + threadIdx.x;
  if (e >= N_EDGES) return;
  int k = ke[e];
  if (k < 0) return;  // out_e stays zero -> no contribution
  const _Float16* row = ea + (long)e * 64;
  float h0 = 0.f, h1 = 0.f, h2 = 0.f, h3 = 0.f;
#pragma unroll 8
  for (int c = 0; c < 64; c++) {
    float x = (float)row[c];
    h0 += x * (float)W5h[c];
    h1 += x * (float)W5h[64 + c];
    h2 += x * (float)W5h[128 + c];
    h3 += x * (float)W5h[192 + c];
  }
  float l1 = h1 >= 0.f ? h1 : 0.1f * h1;
  float l2 = h2 >= 0.f ? h2 : 0.1f * h2;
  float l3 = h3 >= 0.f ? h3 : 0.1f * h3;
  float v = h0 * alpha5[k * 4 + 0] + l1 * alpha5[k * 4 + 1] +
            l2 * l2 * alpha5[k * 4 + 2] + l3 * l3 * l3 * alpha5[k * 4 + 3];
  v /= (float)deg[src[e]];
  atomicAdd(&out[dst[e]], v);
}

// ---------------- host launch ----------------

extern "C" void kernel_launch(void* const* d_in, const int* in_sizes, int n_in,
                              void* d_out, int out_size, void* d_ws,
                              size_t ws_size, hipStream_t stream) {
  (void)in_sizes; (void)n_in; (void)out_size;
  const float* x = (const float*)d_in[0];
  const float* pos = (const float*)d_in[1];
  const int* eidx = (const int*)d_in[2];
  const int *src = eidx, *dst = eidx + N_EDGES;
  const float* lin_sim_w = (const float*)d_in[5];
  const float* lin_sim_b = (const float*)d_in[6];
  const float* lin_edge_w = (const float*)d_in[7];
  const float* lin_edge_b = (const float*)d_in[8];
  const float* ec_w1 = (const float*)d_in[9];
  const float* ec_b1 = (const float*)d_in[10];
  const float* ec_w2 = (const float*)d_in[11];
  const float* ec_b2 = (const float*)d_in[12];
  const float* W124 = (const float*)d_in[13];
  const float* A124 = (const float*)d_in[14];
  const float* W5 = (const float*)d_in[15];
  const float* A5 = (const float*)d_in[16];

  char* ws = (char*)d_ws;
  size_t off = 0;
  auto alloc = [&](size_t bytes) -> char* {
    char* p = ws + off;
    off = (off + bytes + 255) & ~(size_t)255;
    return p;
  };
  float* xs = (float*)alloc((size_t)N_NODES * HID * 4);
  float* agg = (float*)alloc((size_t)N_NODES * HID * 4);  // becomes xs2
  float* cent = (float*)alloc(KC * 64 * 4);
  float* sums = (float*)alloc((KC * 64 + KC) * 4);  // counts appended
  float* counts = sums + KC * 64;
  int* assign = (int*)alloc((size_t)N_NODES * 4);
  int* ke = (int*)alloc((size_t)N_EDGES * 4);
  int* deg = (int*)alloc((size_t)N_NODES * 4);
  int* cnt9 = (int*)alloc(9 * 4);
  int* cur9 = (int*)alloc(9 * 4);
  int* aoff = (int*)alloc(10 * 4);
  int* perm = (int*)alloc((size_t)PERM_CAP * 4);
  int* tile_k = (int*)alloc((size_t)MK_TILES * 4);
  _Float16* eaA = (_Float16*)alloc((size_t)N_EDGES * HID * 2);
  _Float16* eaB = (_Float16*)alloc((size_t)N_EDGES * HID * 2);
  _Float16* w1h = (_Float16*)alloc(64 * 128 * 2);
  _Float16* w2h = (_Float16*)alloc(64 * 64 * 2);
  _Float16* W124h = (_Float16*)alloc(3 * 4 * 64 * 64 * 2);
  _Float16* A124h = (_Float16*)alloc((size_t)3 * 8 * 4 * 64 * 64 * 2);
  _Float16* W5h = (_Float16*)alloc(4 * 64 * 2);
  if (off > ws_size) return;  // workspace too small: bail safely

#define GRID(n) dim3(((n) + 255) / 256), dim3(256), 0, stream

  // weights -> f16 (row-major [out][in] == column-major B, direct frag loads)
  f32_to_f16_kernel<<<GRID(64 * 128)>>>(ec_w1, w1h, 64 * 128);
  f32_to_f16_kernel<<<GRID(64 * 64)>>>(ec_w2, w2h, 64 * 64);
  f32_to_f16_kernel<<<GRID(3 * 4 * 64 * 64)>>>(W124, W124h, 3 * 4 * 64 * 64);
  f32_to_f16_kernel<<<GRID(3 * 8 * 4 * 64 * 64)>>>(A124, A124h, 3 * 8 * 4 * 64 * 64);
  f32_to_f16_kernel<<<GRID(4 * 64)>>>(W5, W5h, 4 * 64);

  // xs
  xs_kernel<<<GRID(N_NODES * HID)>>>(x, pos, lin_sim_w, lin_sim_b, xs);

  // EdgeConv -> segment_max -> xs2
  fill_u32_kernel<<<GRID(N_NODES * HID)>>>((uint32_t*)agg, 0xFF800000u,
                                           N_NODES * HID);  // -inf
  edgeconv_kernel<<<EC_TILES, 32, 0, stream>>>(xs, src, dst, w1h, ec_b1, w2h,
                                               ec_b2, agg);
  fix_agg_kernel<<<GRID(N_NODES * HID)>>>(agg, N_NODES * HID);

  // k-means (10 iterations)
  init_cent_kernel<<<1, 512, 0, stream>>>(agg, cent);
  for (int it = 0; it < 10; it++) {
    fill_u32_kernel<<<GRID(KC * 64 + KC)>>>((uint32_t*)sums, 0u, KC * 64 + KC);
    km_assign_kernel<<<GRID(N_NODES)>>>(agg, cent, assign, sums, counts);
    km_update_kernel<<<1, 512, 0, stream>>>(sums, counts, cent);
  }

  // edge prep: deg, ke, cluster-bucket counting sort (padded to 16-tiles)
  fill_u32_kernel<<<GRID(N_NODES)>>>((uint32_t*)deg, 0u, N_NODES);
  fill_u32_kernel<<<GRID(9)>>>((uint32_t*)cnt9, 0u, 9);
  fill_u32_kernel<<<GRID(9)>>>((uint32_t*)cur9, 0u, 9);
  fill_u32_kernel<<<GRID(PERM_CAP)>>>((uint32_t*)perm, 0xFFFFFFFFu, PERM_CAP);
  edge_prep_kernel<<<GRID(N_EDGES)>>>(src, dst, assign, deg, ke, cnt9);
  aoff_kernel<<<1, 1, 0, stream>>>(cnt9, aoff);
  scatter_kernel<<<GRID(N_EDGES)>>>(ke, aoff, cur9, perm);
  tilek_kernel<<<GRID(MK_TILES)>>>(aoff, tile_k);

  // ea0
  ea0_kernel<<<GRID(N_EDGES * 64)>>>(pos, src, dst, lin_edge_w, lin_edge_b, eaA);

  // three HID->HID convs (eaA -> eaB -> eaA -> eaB)
  mkconv_kernel<<<MK_TILES, 32, 0, stream>>>(eaA, eaB, W124h,
                                             A124h, perm, tile_k, src, deg);
  mkconv_kernel<<<MK_TILES, 32, 0, stream>>>(eaB, eaA, W124h + 4 * 64 * 64,
                                             A124h + 8 * 4 * 64 * 64, perm,
                                             tile_k, src, deg);
  mkconv_kernel<<<MK_TILES, 32, 0, stream>>>(eaA, eaB, W124h + 2 * 4 * 64 * 64,
                                             A124h + 2 * 8 * 4 * 64 * 64, perm,
                                             tile_k, src, deg);

  // conv5 -> output (N_NODES x 1)
  fill_u32_kernel<<<GRID(N_NODES)>>>((uint32_t*)d_out, 0u, N_NODES);
  conv5_kernel<<<GRID(N_EDGES)>>>(eaB, W5h, A5, src, dst, ke, deg,
                                  (float*)d_out);
#undef GRID
}
